// Mamba2_53652731462178
// MI455X (gfx1250) — matrix operand order repaired and checked by reading, not compiled, not run on hardware
//
#include <hip/hip_runtime.h>
#include <hip/hip_bf16.h>
#include <math.h>

// ---------------------------------------------------------------------------
// Mamba2 block for MI455X (gfx1250): bf16 WMMA for all matmuls, fp32 elsewhere.
// Round 5: async global->LDS staging via GLOBAL_LOAD_ASYNC_TO_LDS_B128 with
// the correct v4i pointer types (per hipcc diagnostic), ASYNCcnt fencing.
// ---------------------------------------------------------------------------

typedef __attribute__((ext_vector_type(16))) __bf16          v16bf;
typedef __attribute__((ext_vector_type(8)))  float           v8f;
typedef __attribute__((ext_vector_type(8)))  unsigned short  us8;
typedef __attribute__((ext_vector_type(16))) unsigned short  us16;
typedef __attribute__((ext_vector_type(4)))  int             v4i;

#define BSZ_   2
#define LSEQ_  4096
#define DMODEL 1024
#define DI_    2048
#define DPRJ_  4384
#define DPRJP_ 4480   // padded to 35*128 so GEMM1 has no ragged N
#define CVD_   2304
#define NH_    32
#define HP_    64
#define DS_    128
#define CS_    64

#if defined(__has_builtin)
#if __has_builtin(__builtin_amdgcn_global_load_async_to_lds_b128)
#define HAVE_ASYNC_LDS 1
#endif
#endif
#ifndef HAVE_ASYNC_LDS
#define HAVE_ASYNC_LDS 0
#endif

typedef __attribute__((address_space(1))) v4i* gptr_v4i;
typedef __attribute__((address_space(3))) v4i* lptr_v4i;

// ---------------- bf16 helpers ----------------
static __device__ __forceinline__ unsigned short f2bfbits(float f) {
  union { float f; unsigned u; } v; v.f = f;
  unsigned r = v.u + 0x7FFFu + ((v.u >> 16) & 1u);   // round-to-nearest-even
  return (unsigned short)(r >> 16);
}
static __device__ __forceinline__ float bf2f(unsigned short s) {
  union { unsigned u; float f; } v; v.u = ((unsigned)s) << 16; return v.f;
}

static __device__ __forceinline__ v8f zero8() {
  v8f z;
#pragma unroll
  for (int i = 0; i < 8; ++i) z[i] = 0.0f;
  return z;
}

static __device__ __forceinline__ v8f wmma_bf16(v16bf a, v16bf b, v8f c) {
  return __builtin_amdgcn_wmma_f32_16x16x32_bf16(
      /*neg_a=*/false, a, /*neg_b=*/false, b,
      /*c_mod=*/(short)0, c, /*reuse_a=*/false, /*reuse_b=*/false);
}

// 16B global->LDS copy: async on gfx1250 when available, sync vector otherwise.
static __device__ __forceinline__ void copy16(const unsigned short* g,
                                              unsigned short* l) {
#if HAVE_ASYNC_LDS
  __builtin_amdgcn_global_load_async_to_lds_b128(
      (gptr_v4i)(g), (lptr_v4i)(l), 0, 0);
#else
  *(us8*)__builtin_assume_aligned(l, 16) =
      *(const us8*)__builtin_assume_aligned(g, 16);
#endif
}
// Drain outstanding async copies issued by this wave.
static __device__ __forceinline__ void copy_fence() {
#if HAVE_ASYNC_LDS
#if defined(__has_builtin) && __has_builtin(__builtin_amdgcn_s_wait_asynccnt)
  __builtin_amdgcn_s_wait_asynccnt(0);
#else
  asm volatile("s_wait_asynccnt 0x0" ::: "memory");
#endif
#endif
}

// Build a 16-element bf16 fragment from two aligned 8-element (16B) LDS runs.
static __device__ __forceinline__ v16bf frag2(const unsigned short* p0,
                                              const unsigned short* p1) {
  const us8* q0 = (const us8*)__builtin_assume_aligned(p0, 16);
  const us8* q1 = (const us8*)__builtin_assume_aligned(p1, 16);
  us8 a = *q0, b = *q1;
  us16 w = __builtin_shufflevector(a, b, 0, 1, 2, 3, 4, 5, 6, 7,
                                         8, 9, 10, 11, 12, 13, 14, 15);
  return __builtin_bit_cast(v16bf, w);
}

// A fragment (16x32), row-major src[m][k], ld % 8 == 0, k0 % 8 == 0.
static __device__ __forceinline__ v16bf load_a_v(const unsigned short* src,
                                                 int ld, int row0, int k0) {
  const int lane = threadIdx.x & 31;
  const int h = lane >> 4, m = lane & 15;
  const unsigned short* base = src + (row0 + m) * ld + k0 + h * 8;
  return frag2(base, base + 16);
}
// B fragment (32x16) from TRANSPOSED storage srcT[n][k], ld % 8 == 0.
static __device__ __forceinline__ v16bf load_b_v(const unsigned short* srcT,
                                                 int ld, int k0, int col0) {
  const int lane = threadIdx.x & 31;
  const int h = lane >> 4, n = lane & 15;
  const unsigned short* base = srcT + (col0 + n) * ld + k0 + h * 8;
  return frag2(base, base + 16);
}

// ---------------- fp32 -> bf16 convert (flat) ----------------
__global__ void f32_to_bf16_kernel(const float* __restrict__ in,
                                   unsigned short* __restrict__ out, long n) {
  long i = (long)blockIdx.x * 256 + threadIdx.x;
  if (i < n) out[i] = f2bfbits(in[i]);
}

// ---------------- fp32 [K][N] -> bf16 transposed+padded [Npad][K] ----------------
__global__ void f32_to_bf16_tpad_kernel(const float* __restrict__ in,
                                        unsigned short* __restrict__ out,
                                        int K, int N, int Npad) {
  long idx = (long)blockIdx.x * 256 + threadIdx.x;
  long total = (long)Npad * K;
  if (idx >= total) return;
  int k = (int)(idx % K);
  int n = (int)(idx / K);
  out[idx] = (n < N) ? f2bfbits(in[(long)k * N + n]) : (unsigned short)0;
}

// ---------------- big bf16 GEMM: C[MxN] f32 = A[MxK] * BT[NxK]^T ----------------
// Launcher guarantees M % 128 == 0, N % 128 == 0, K % 32 == 0.
#define GT_M 128
#define GT_N 128
#define GT_K 32
#define GT_LD 40   // 32 + 8: keeps every 8-element run 16B-aligned

__global__ void __launch_bounds__(256)
gemm_bf16_kernel(const unsigned short* __restrict__ A,   // [M][K]
                 const unsigned short* __restrict__ BT,  // [N][K]
                 float* __restrict__ C, int M, int N, int K) {
  __shared__ __align__(16) unsigned short sA[2][GT_M][GT_LD];  // [m][k]
  __shared__ __align__(16) unsigned short sB[2][GT_N][GT_LD];  // [n][k]

  const int tid  = threadIdx.x;
  const int wave = tid >> 5;
  const int m0 = blockIdx.y * GT_M;
  const int n0 = blockIdx.x * GT_N;
  const int wm = (wave & 3) * 32;   // wave row offset in tile
  const int wn = (wave >> 2) * 64;  // wave col offset in tile

  // staging coordinates: 512 8-elem runs per operand tile, 2 per thread
  const int ar0 = (tid * 2) >> 2,       ac0 = ((tid * 2) & 3) * 8;
  const int ar1 = (tid * 2 + 1) >> 2,   ac1 = ((tid * 2 + 1) & 3) * 8;

  v8f acc[2][4];
#pragma unroll
  for (int i = 0; i < 2; ++i)
#pragma unroll
    for (int j = 0; j < 4; ++j) acc[i][j] = zero8();

  auto stage_tile = [&](int kk, int buf) {
    copy16(&A [(long)(m0 + ar0) * K + kk + ac0], &sA[buf][ar0][ac0]);
    copy16(&A [(long)(m0 + ar1) * K + kk + ac1], &sA[buf][ar1][ac1]);
    copy16(&BT[(long)(n0 + ar0) * K + kk + ac0], &sB[buf][ar0][ac0]);
    copy16(&BT[(long)(n0 + ar1) * K + kk + ac1], &sB[buf][ar1][ac1]);
  };

  const int T = K / GT_K;
  stage_tile(0, 0);
  copy_fence();
  __syncthreads();

  for (int t = 0; t < T; ++t) {
    const int cur = t & 1;
    const bool more = (t + 1) < T;
    if (more) stage_tile((t + 1) * GT_K, cur ^ 1);  // overlaps WMMA below

    v16bf a0 = load_a_v(&sA[cur][0][0], GT_LD, wm, 0);
    v16bf a1 = load_a_v(&sA[cur][0][0], GT_LD, wm + 16, 0);
#pragma unroll
    for (int j = 0; j < 4; ++j) {
      v16bf bj = load_b_v(&sB[cur][0][0], GT_LD, 0, wn + j * 16);
      acc[0][j] = wmma_bf16(a0, bj, acc[0][j]);
      acc[1][j] = wmma_bf16(a1, bj, acc[1][j]);
    }
    if (more) copy_fence();
    __syncthreads();
  }

  const int lane = tid & 31, h = lane >> 4, n = lane & 15;
#pragma unroll
  for (int i = 0; i < 2; ++i)
#pragma unroll
    for (int j = 0; j < 4; ++j)
#pragma unroll
      for (int v = 0; v < 8; ++v) {
        int r = m0 + wm + i * 16 + v + 8 * h;
        int c = n0 + wn + j * 16 + n;
        C[(long)r * N + c] = acc[i][j][v];
      }
}

// ---------------- causal depthwise conv (k=4) + SiLU -> bf16 ----------------
__global__ void conv_silu_kernel(const float* __restrict__ zxbcdt,  // [rows][DPRJP_]
                                 const float* __restrict__ cw,
                                 const float* __restrict__ cb,
                                 unsigned short* __restrict__ xBCc, long total) {
  long idx = (long)blockIdx.x * 256 + threadIdx.x;
  if (idx >= total) return;
  int ch = (int)(idx % CVD_);
  long t = idx / CVD_;
  int tl = (int)(t % LSEQ_);
  long bb = t / LSEQ_;
  float acc = cb[ch];
#pragma unroll
  for (int i = 0; i < 4; ++i) {
    int ts = tl - 3 + i;
    if (ts >= 0)
      acc += zxbcdt[(bb * LSEQ_ + ts) * (long)DPRJP_ + DI_ + ch] * cw[ch * 4 + i];
  }
  xBCc[idx] = f2bfbits(acc / (1.0f + expf(-acc)));  // SiLU, bf16 out
}

// ---------------- dt = softplus(raw + bias) ----------------
__global__ void dt_kernel(const float* __restrict__ zxbcdt,
                          const float* __restrict__ dt_bias,
                          float* __restrict__ dtv, long total) {
  long idx = (long)blockIdx.x * 256 + threadIdx.x;
  if (idx >= total) return;
  int h = (int)(idx & (NH_ - 1));
  long row = idx >> 5;
  float x = zxbcdt[row * (long)DPRJP_ + (DPRJ_ - NH_) + h] + dt_bias[h];
  dtv[idx] = (x > 20.0f) ? x : log1pf(expf(x));
}

// ---------------- chunked SSD scan: one block per (batch, head) ----------------
__global__ void __launch_bounds__(256)
ssd_kernel(const unsigned short* __restrict__ xBCc,  // [B*L][2304] bf16
           const float* __restrict__ dtv,            // [B*L][32]
           const float* __restrict__ A_log,          // [32]
           float* __restrict__ y) {                  // [B*L][2048]
  __shared__ float s_state[DS_][HP_ + 1];                         // fp32 state [n][p]
  __shared__ __align__(16) unsigned short s_stateT[HP_][DS_ + 8]; // bf16 [p][n]
  __shared__ __align__(16) unsigned short s_xT[HP_][CS_ + 8];     // bf16 [p][s] raw x
  __shared__ __align__(16) unsigned short s_Bm[CS_][DS_ + 8];     // bf16 [s][n]
  __shared__ __align__(16) unsigned short s_Cm[CS_][DS_ + 8];     // bf16 [s][n]; ->Cexp
  __shared__ __align__(16) unsigned short s_BdT[DS_][CS_ + 8];    // bf16 [n][s] = (Bd*dt)^T
  __shared__ float s_G[CS_][CS_ + 1];                             // fp32 C*B^T
  __shared__ __align__(16) unsigned short s_Gb[CS_][CS_ + 8];     // bf16 masked G * dt
  __shared__ float s_ac[CS_];
  __shared__ float s_av[CS_];
  __shared__ float s_dt[CS_];

  const int tid  = threadIdx.x;
  const int wave = tid >> 5;
  const int b  = blockIdx.x >> 5;
  const int hh = blockIdx.x & 31;
  const float Ah = -expf(A_log[hh]);

  for (int i = tid; i < DS_ * HP_; i += 256) {
    int n = i >> 6, p = i & 63;
    s_state[n][p] = 0.0f;
    s_stateT[p][n] = 0;
  }
  __syncthreads();

  const int nchunks = LSEQ_ / CS_;
  for (int c = 0; c < nchunks; ++c) {
    const int t0 = b * LSEQ_ + c * CS_;

    // ---- stage chunk operands ----
    // B and C: 16B copies (async to LDS when available)
#pragma unroll
    for (int it = 0; it < 4; ++it) {
      int i = tid + it * 256;                 // 0..1023
      int s = i >> 4, cs = (i & 15) * 8;
      copy16(&xBCc[(t0 + s) * (long)CVD_ + DI_ + cs],       &s_Bm[s][cs]);
      copy16(&xBCc[(t0 + s) * (long)CVD_ + DI_ + DS_ + cs], &s_Cm[s][cs]);
    }
    if (tid < CS_) {
      float d = dtv[(t0 + tid) * NH_ + hh];
      s_dt[tid] = d;
      s_av[tid] = Ah * d;
    }
    // x: raw bf16, transposed into s_xT[p][s]
    for (int i = tid; i < CS_ * HP_; i += 256) {
      int s = i >> 6, p = i & 63;
      s_xT[p][s] = xBCc[(t0 + s) * (long)CVD_ + hh * HP_ + p];
    }
    copy_fence();
    __syncthreads();
    if (tid == 0) {
      float a = 0.0f;
      for (int s = 0; s < CS_; ++s) { a += s_av[s]; s_ac[s] = a; }
    }
    __syncthreads();
    const float alast = s_ac[CS_ - 1];

    // ---- phase 1: G = C * B^T  (64x64, K=128) — 16 tiles / 8 waves ----
#pragma unroll
    for (int q = 0; q < 2; ++q) {
      const int tt = wave * 2 + q;
      const int r0 = (tt >> 2) * 16, c0 = (tt & 3) * 16;
      v8f acc = zero8();
#pragma unroll
      for (int k0 = 0; k0 < DS_; k0 += 32) {
        v16bf af = load_a_v(&s_Cm[0][0], DS_ + 8, r0, k0);
        v16bf bf = load_b_v(&s_Bm[0][0], DS_ + 8, k0, c0);  // B[k=n][t'] = Bm[t'][n]
        acc = wmma_bf16(af, bf, acc);
      }
      const int lane = tid & 31, h = lane >> 4, n = lane & 15;
#pragma unroll
      for (int v = 0; v < 8; ++v) s_G[r0 + v + 8 * h][c0 + n] = acc[v];
    }
    __syncthreads();

    // ---- phase 2: mask+decay G (fold dt[s']); C->Cexp; BdT (fold dt[s]) ----
    for (int i = tid; i < CS_ * CS_; i += 256) {
      int s = i >> 6, t2 = i & 63;
      float g = (s >= t2)
                    ? s_G[s][t2] * expf(s_ac[s] - s_ac[t2]) * s_dt[t2]
                    : 0.0f;
      s_Gb[s][t2] = f2bfbits(g);
    }
    for (int i = tid; i < CS_ * DS_; i += 256) {
      int s = i >> 7, n = i & 127;
      s_Cm[s][n] = f2bfbits(bf2f(s_Cm[s][n]) * expf(s_ac[s]));
      s_BdT[n][s] = f2bfbits(bf2f(s_Bm[s][n]) * expf(alast - s_ac[s]) * s_dt[s]);
    }
    __syncthreads();

    // ---- phase 3: Y = Gm*x + Cexp*state  (64x64) ----
#pragma unroll
    for (int q = 0; q < 2; ++q) {
      const int tt = wave * 2 + q;
      const int r0 = (tt >> 2) * 16, c0 = (tt & 3) * 16;
      v8f acc = zero8();
#pragma unroll
      for (int k0 = 0; k0 < CS_; k0 += 32) {      // diag part, K = 64
        v16bf af = load_a_v(&s_Gb[0][0], CS_ + 8, r0, k0);
        v16bf bf = load_b_v(&s_xT[0][0], CS_ + 8, k0, c0);
        acc = wmma_bf16(af, bf, acc);
      }
#pragma unroll
      for (int k0 = 0; k0 < DS_; k0 += 32) {      // off-diag part, K = 128
        v16bf af = load_a_v(&s_Cm[0][0], DS_ + 8, r0, k0);
        v16bf bf = load_b_v(&s_stateT[0][0], DS_ + 8, k0, c0);
        acc = wmma_bf16(af, bf, acc);
      }
      const int lane = tid & 31, h = lane >> 4, n = lane & 15;
#pragma unroll
      for (int v = 0; v < 8; ++v) {
        int s = r0 + v + 8 * h, p = c0 + n;
        y[(t0 + s) * (long)DI_ + hh * HP_ + p] = acc[v];
      }
    }
    __syncthreads();

    // ---- phase 4: state = exp(alast)*state + BdT * x  (128x64, K=64) ----
    {
      const float edec = expf(alast);
      const int r0 = wave * 16;
      v8f acc4[4];
#pragma unroll
      for (int j = 0; j < 4; ++j) acc4[j] = zero8();
#pragma unroll
      for (int k0 = 0; k0 < CS_; k0 += 32) {
        v16bf af = load_a_v(&s_BdT[0][0], CS_ + 8, r0, k0);
#pragma unroll
        for (int j = 0; j < 4; ++j) {
          v16bf bf = load_b_v(&s_xT[0][0], CS_ + 8, k0, j * 16);
          acc4[j] = wmma_bf16(af, bf, acc4[j]);
        }
      }
      const int lane = tid & 31, h = lane >> 4, n = lane & 15;
#pragma unroll
      for (int j = 0; j < 4; ++j)
#pragma unroll
        for (int v = 0; v < 8; ++v) {
          int nr = r0 + v + 8 * h, p = j * 16 + n;
          float nv = edec * s_state[nr][p] + acc4[j][v];
          s_state[nr][p] = nv;
          s_stateT[p][nr] = f2bfbits(nv);
        }
    }
    __syncthreads();
  }
}

// ---------------- gated norm: LN(y * silu(z)) -> bf16 ----------------
__global__ void __launch_bounds__(256)
gatenorm_kernel(const float* __restrict__ yin,     // [rows][2048]
                const float* __restrict__ zxbcdt,  // [rows][DPRJP_]
                const float* __restrict__ nw, const float* __restrict__ nb,
                unsigned short* __restrict__ ybf) {
  __shared__ float red[256];
  const int row = blockIdx.x;
  const int tid = threadIdx.x;
  float v[8];
  float s1 = 0.0f;
#pragma unroll
  for (int i = 0; i < 8; ++i) {
    int ch = tid + i * 256;
    float z = zxbcdt[(long)row * DPRJP_ + ch];
    float val = yin[(long)row * DI_ + ch] * (z / (1.0f + expf(-z)));
    v[i] = val; s1 += val;
  }
  red[tid] = s1; __syncthreads();
  for (int s = 128; s > 0; s >>= 1) { if (tid < s) red[tid] += red[tid + s]; __syncthreads(); }
  const float mu = red[0] / (float)DI_;
  __syncthreads();
  float s2 = 0.0f;
#pragma unroll
  for (int i = 0; i < 8; ++i) { float d = v[i] - mu; s2 += d * d; }
  red[tid] = s2; __syncthreads();
  for (int s = 128; s > 0; s >>= 1) { if (tid < s) red[tid] += red[tid + s]; __syncthreads(); }
  const float rs = rsqrtf(red[0] / (float)DI_ + 1e-5f);
#pragma unroll
  for (int i = 0; i < 8; ++i) {
    int ch = tid + i * 256;
    ybf[(long)row * DI_ + ch] = f2bfbits((v[i] - mu) * rs * nw[ch] + nb[ch]);
  }
}

// ---------------------------------------------------------------------------
extern "C" void kernel_launch(void* const* d_in, const int* in_sizes, int n_in,
                              void* d_out, int out_size, void* d_ws, size_t ws_size,
                              hipStream_t stream) {
  const float* u       = (const float*)d_in[0];
  const float* W_in    = (const float*)d_in[1];
  const float* conv_w  = (const float*)d_in[2];
  const float* conv_b  = (const float*)d_in[3];
  const float* dt_bias = (const float*)d_in[4];
  const float* A_log   = (const float*)d_in[5];
  const float* norm_w  = (const float*)d_in[6];
  const float* norm_b  = (const float*)d_in[7];
  const float* W_out   = (const float*)d_in[8];
  float* out = (float*)d_out;

  const long rows = (long)BSZ_ * LSEQ_;  // 8192

  char* ws = (char*)d_ws;
  auto alloc = [&](size_t bytes) -> char* {
    char* p = ws;
    ws += (bytes + 255) & ~(size_t)255;
    return p;
  };
  unsigned short* u_bf     = (unsigned short*)alloc((size_t)rows * DMODEL * 2);
  unsigned short* WinT_bf  = (unsigned short*)alloc((size_t)DPRJP_ * DMODEL * 2); // [Npad][K]
  unsigned short* WoutT_bf = (unsigned short*)alloc((size_t)DMODEL * DI_ * 2);    // [N][K]
  float* zxbcdt = (float*)alloc((size_t)rows * DPRJP_ * 4);
  unsigned short* xBCc = (unsigned short*)alloc((size_t)rows * CVD_ * 2);
  float* dtv    = (float*)alloc((size_t)rows * NH_ * 4);
  float* yint   = (float*)alloc((size_t)rows * DI_ * 4);
  unsigned short* ybf = (unsigned short*)alloc((size_t)rows * DI_ * 2);

  // 1) weight/activation conversions (weights transposed, W_in zero-padded)
  {
    long n = rows * DMODEL;
    f32_to_bf16_kernel<<<(unsigned)((n + 255) / 256), 256, 0, stream>>>(u, u_bf, n);
    long nt = (long)DPRJP_ * DMODEL;
    f32_to_bf16_tpad_kernel<<<(unsigned)((nt + 255) / 256), 256, 0, stream>>>(
        W_in, WinT_bf, DMODEL, DPRJ_, DPRJP_);
    nt = (long)DMODEL * DI_;
    f32_to_bf16_tpad_kernel<<<(unsigned)((nt + 255) / 256), 256, 0, stream>>>(
        W_out, WoutT_bf, DI_, DMODEL, DMODEL);
  }

  // 2) in_proj GEMM: zxbcdt[8192 x 4480] = u[8192 x 1024] @ W_in (padded)
  {
    dim3 g(DPRJP_ / GT_N, (unsigned)(rows / GT_M));
    gemm_bf16_kernel<<<g, 256, 0, stream>>>(u_bf, WinT_bf, zxbcdt,
                                            (int)rows, DPRJP_, DMODEL);
  }

  // 3) causal dwconv + SiLU -> bf16
  {
    long n = rows * (long)CVD_;
    conv_silu_kernel<<<(unsigned)((n + 255) / 256), 256, 0, stream>>>(
        zxbcdt, conv_w, conv_b, xBCc, n);
  }

  // 4) dt = softplus(raw + bias)
  {
    long n = rows * NH_;
    dt_kernel<<<(unsigned)((n + 255) / 256), 256, 0, stream>>>(zxbcdt, dt_bias, dtv, n);
  }

  // 5) chunked SSD scan, one block per (batch, head)
  ssd_kernel<<<BSZ_ * NH_, 256, 0, stream>>>(xBCc, dtv, A_log, yint);

  // 6) gated LayerNorm -> bf16
  gatenorm_kernel<<<(unsigned)rows, 256, 0, stream>>>(yint, zxbcdt, norm_w, norm_b, ybf);

  // 7) out_proj GEMM: out[8192 x 1024] = y[8192 x 2048] @ W_out
  {
    dim3 g(DMODEL / GT_N, (unsigned)(rows / GT_M));
    gemm_bf16_kernel<<<g, 256, 0, stream>>>(ybf, WoutT_bf, out,
                                            (int)rows, DMODEL, DI_);
  }
}
